// MultiHeadAttentionLayer_73418170958216
// MI455X (gfx1250) — matrix operand order repaired and checked by reading, not compiled, not run on hardware
//
#include <hip/hip_runtime.h>
#include <hip/hip_bf16.h>
#include <math.h>

#define NODES_DIM 512
#define KNN_K 16
#define TILE_N 64
#define BLOCK_M 256
#define LDS_ROW 520   // padded bf16 row: 512 data + 8 pad (1040 B, distinct LDS banks)

typedef float v8f  __attribute__((ext_vector_type(8)));
typedef __bf16 v16bf __attribute__((ext_vector_type(16)));
typedef unsigned int v4u __attribute__((ext_vector_type(4)));
typedef int v8i __attribute__((ext_vector_type(8)));
typedef int v4i __attribute__((ext_vector_type(4)));

// ---------------------------------------------------------------------------
// fp32 -> bf16 hi/lo split (hi = rne(x), lo = rne(x - hi)); exact fp32 exponent
// coverage since bf16 shares the fp32 exponent field.
// ---------------------------------------------------------------------------
__global__ __launch_bounds__(256)
void split_bf16_kernel(const float* __restrict__ x,
                       __bf16* __restrict__ hi, __bf16* __restrict__ lo, int n)
{
    int i = blockIdx.x * blockDim.x + threadIdx.x;
    if (i >= n) return;
    float v = x[i];
    __bf16 h = (__bf16)v;
    float r = v - (float)h;
    hi[i] = h;
    lo[i] = (__bf16)r;
}

// ---------------------------------------------------------------------------
// TDM helper: DMA 8 rows x 512 bf16 from global (row stride 512 elts) into LDS
// with 16B padding inserted after every 1024B row -> padded LDS row = 1040 B.
// D# built per CDNA5 ISA §8.3/8.4 (group0: count/lds/global/type,
// group1: data_size=2B, pad_enable, pad_interval=256 dwords, pad_amount=4 dwords,
// dims/strides; groups 2-3 zero for a 2D tile).
// ---------------------------------------------------------------------------
__device__ __forceinline__ void tdm_load_rows8(unsigned long long ga, unsigned lds_addr)
{
    v4u g0;
    g0[0] = 1u;                                               // count = 1, no gather
    g0[1] = lds_addr;                                         // LDS byte address
    g0[2] = (unsigned)(ga & 0xFFFFFFFFu);                     // global_addr[31:0]
    g0[3] = (unsigned)((ga >> 32) & 0x01FFFFFFu) | (2u << 30);// global_addr[56:32] | type=2
    v8i g1;
    g1[0] = (int)((1u << 16)      // data_size = 1 -> 2 bytes
                | (1u << 20)      // pad_enable
                | (7u << 22)      // pad_interval code 7 -> 256 dwords (1024 B)
                | (3u << 25));    // pad_amount  code 3 -> 4 dwords (16 B)
    g1[1] = (int)(512u << 16);    // tensor_dim0[15:0] << 16
    g1[2] = (int)(512u << 16);    // tensor_dim0[31:16]=0 | tensor_dim1[15:0] << 16
    g1[3] = (int)(512u << 16);    // tensor_dim1[31:16]=0 | tile_dim0 = 512
    g1[4] = 8;                    // tile_dim1 = 8 rows, tile_dim2 = 0
    g1[5] = 512;                  // tensor_dim0_stride (elements)
    g1[6] = 0;
    g1[7] = 0;
    v4i z4 = {0, 0, 0, 0};
    v8i z8 = {0, 0, 0, 0, 0, 0, 0, 0};
    __builtin_amdgcn_tensor_load_to_lds(g0, g1, z4, z4, z8, 0);
}

// assemble a 16-element bf16 fragment from two 16-byte runs
__device__ __forceinline__ v16bf load_frag(const __bf16* p0, const __bf16* p1)
{
    union { v16bf v; v4u u[2]; } f;
    f.u[0] = *(const v4u*)p0;
    f.u[1] = *(const v4u*)p1;
    return f.v;
}

// ---------------------------------------------------------------------------
// Kernel 1: Q/K/V projections, bf16x3 split-precision WMMA (K=32 per op).
//   C[i][j] = dot(h[i], W[j])  via  Ahi*Bhi + Ahi*Blo + Alo*Bhi  (fp32 accum).
//   Block = 256 thr = 8 waves; wave tile = 32M x 64N (2 M-tiles, 4 N-tiles).
//   B tile (64 cols x 512 K, hi+lo) staged in LDS by the Tensor Data Mover.
// ---------------------------------------------------------------------------
__global__ __launch_bounds__(256)
void qkv_gemm_bf16x3(const __bf16* __restrict__ h_hi, const __bf16* __restrict__ h_lo,
                     const __bf16* __restrict__ Wqh, const __bf16* __restrict__ Wql,
                     const __bf16* __restrict__ Wkh, const __bf16* __restrict__ Wkl,
                     const __bf16* __restrict__ Wvh, const __bf16* __restrict__ Wvl,
                     float* __restrict__ Qo, float* __restrict__ Ko, float* __restrict__ Vo)
{
    extern __shared__ __bf16 sB[];   // [hi|lo][64][LDS_ROW]

    const __bf16 *Wh, *Wl;
    float* C;
    if (blockIdx.z == 0)      { Wh = Wqh; Wl = Wql; C = Qo; }
    else if (blockIdx.z == 1) { Wh = Wkh; Wl = Wkl; C = Ko; }
    else                      { Wh = Wvh; Wl = Wvl; C = Vo; }

    const int wave = threadIdx.x >> 5;
    const int lane = threadIdx.x & 31;
    const int half = lane >> 4;
    const int l16  = lane & 15;
    const int nbase = blockIdx.y * TILE_N;
    const int mbase = blockIdx.x * BLOCK_M + wave * 32;

    // --- TDM stage of this block's weight tile into LDS (8 rows per wave) ---
    {
        unsigned ldsBase = (unsigned)(unsigned long long)(void*)sB;
        unsigned long long gaH =
            (unsigned long long)(const void*)(Wh + (size_t)(nbase + 8 * wave) * NODES_DIM);
        unsigned long long gaL =
            (unsigned long long)(const void*)(Wl + (size_t)(nbase + 8 * wave) * NODES_DIM);
        tdm_load_rows8(gaH, ldsBase + (unsigned)(wave * 8 * LDS_ROW * 2));
        tdm_load_rows8(gaL, ldsBase + (unsigned)((64 + wave * 8) * LDS_ROW * 2));
        __builtin_amdgcn_s_wait_tensorcnt(0);
    }
    __syncthreads();

    v8f acc[2][4];
#pragma unroll
    for (int mt = 0; mt < 2; ++mt)
#pragma unroll
        for (int t = 0; t < 4; ++t) acc[mt][t] = (v8f){};

    const __bf16* a0h = h_hi + (size_t)(mbase + l16) * NODES_DIM;
    const __bf16* a0l = h_lo + (size_t)(mbase + l16) * NODES_DIM;
    const __bf16* a1h = h_hi + (size_t)(mbase + 16 + l16) * NODES_DIM;
    const __bf16* a1l = h_lo + (size_t)(mbase + 16 + l16) * NODES_DIM;

    for (int k = 0; k < NODES_DIM; k += 32) {
        // bf16 A 16x32 layout: lanes 0-15 hold K {k..k+7, k+16..k+23},
        //                      lanes 16-31 hold K {k+8..k+15, k+24..k+31}
        const int e0 = k + 8 * half;
        const int e1 = k + 16 + 8 * half;
        if (k + 128 < NODES_DIM) __builtin_prefetch(a0h + k + 128, 0, 3);
        v16bf A0h = load_frag(a0h + e0, a0h + e1);
        v16bf A0l = load_frag(a0l + e0, a0l + e1);
        v16bf A1h = load_frag(a1h + e0, a1h + e1);
        v16bf A1l = load_frag(a1l + e0, a1l + e1);

#pragma unroll
        for (int t = 0; t < 4; ++t) {
            // bf16 B 32x16 layout: lane l16 -> column 16t+l16,
            // lanes 0-15: K {k..k+7, k+16..k+23}, lanes 16-31: K {k+8..k+15, k+24..k+31}
            const __bf16* bh = sB + (size_t)(16 * t + l16) * LDS_ROW + (k + 8 * half);
            const __bf16* bl = bh + (size_t)64 * LDS_ROW;
            v16bf Bh = load_frag(bh, bh + 16);
            v16bf Bl = load_frag(bl, bl + 16);

            acc[0][t] = __builtin_amdgcn_wmma_f32_16x16x32_bf16(false, A0h, false, Bh, (short)0, acc[0][t], false, false);
            acc[0][t] = __builtin_amdgcn_wmma_f32_16x16x32_bf16(false, A0h, false, Bl, (short)0, acc[0][t], false, false);
            acc[0][t] = __builtin_amdgcn_wmma_f32_16x16x32_bf16(false, A0l, false, Bh, (short)0, acc[0][t], false, false);
            acc[1][t] = __builtin_amdgcn_wmma_f32_16x16x32_bf16(false, A1h, false, Bh, (short)0, acc[1][t], false, false);
            acc[1][t] = __builtin_amdgcn_wmma_f32_16x16x32_bf16(false, A1h, false, Bl, (short)0, acc[1][t], false, false);
            acc[1][t] = __builtin_amdgcn_wmma_f32_16x16x32_bf16(false, A1l, false, Bh, (short)0, acc[1][t], false, false);
        }
    }

    // C/D layout: VGPR r -> lanes 0-15: M=r, lanes 16-31: M=r+8
#pragma unroll
    for (int mt = 0; mt < 2; ++mt) {
#pragma unroll
        for (int r = 0; r < 8; ++r) {
            int row = mbase + 16 * mt + r + 8 * half;
            float* crow = C + (size_t)row * NODES_DIM + nbase + l16;
            crow[0]  = acc[mt][0][r];
            crow[16] = acc[mt][1][r];
            crow[32] = acc[mt][2][r];
            crow[48] = acc[mt][3][r];
        }
    }
}

// ---------------------------------------------------------------------------
// Kernel 2: G = h @ Wg^T   (Wg is [3, 512]) — fp32, exact
// ---------------------------------------------------------------------------
__global__ __launch_bounds__(256)
void g_proj_kernel(const float* __restrict__ h,
                   const float* __restrict__ Wg,
                   float* __restrict__ G, int n)
{
    int i = blockIdx.x * blockDim.x + threadIdx.x;
    if (i >= n) return;
    const float4* hr = (const float4*)(h + (size_t)i * NODES_DIM);
    const float4* w0 = (const float4*)(Wg);
    const float4* w1 = (const float4*)(Wg + NODES_DIM);
    const float4* w2 = (const float4*)(Wg + 2 * NODES_DIM);
    float s0 = 0.f, s1 = 0.f, s2 = 0.f;
    for (int t = 0; t < NODES_DIM / 4; ++t) {
        float4 x = hr[t];
        float4 a = w0[t], b = w1[t], c = w2[t];
        s0 += x.x * a.x + x.y * a.y + x.z * a.z + x.w * a.w;
        s1 += x.x * b.x + x.y * b.y + x.z * b.z + x.w * b.w;
        s2 += x.x * c.x + x.y * c.y + x.z * c.z + x.w * c.w;
    }
    G[3 * i + 0] = s0;
    G[3 * i + 1] = s1;
    G[3 * i + 2] = s2;
}

// ---------------------------------------------------------------------------
// Kernel 3: KNN top-16. All 8192 float3 coords (96 KB) staged in LDS.
// ---------------------------------------------------------------------------
__global__ __launch_bounds__(256)
void knn_kernel(const float* __restrict__ G, int* __restrict__ nbr, int n)
{
    extern __shared__ float sc[];
    for (int t = threadIdx.x; t < 3 * n; t += blockDim.x) sc[t] = G[t];
    __syncthreads();

    int i = blockIdx.x * blockDim.x + threadIdx.x;
    if (i >= n) return;

    float xi = sc[3 * i], yi = sc[3 * i + 1], zi = sc[3 * i + 2];

    float bd[KNN_K];
    int   bi[KNN_K];
#pragma unroll
    for (int p = 0; p < KNN_K; ++p) { bd[p] = 3.4e38f; bi[p] = -1; }

    for (int j = 0; j < n; ++j) {
        float dx = sc[3 * j]     - xi;
        float dy = sc[3 * j + 1] - yi;
        float dz = sc[3 * j + 2] - zi;
        float d2 = dx * dx + dy * dy + dz * dz;
        if (d2 < bd[KNN_K - 1]) {
            float dc = d2; int ic = j;
#pragma unroll
            for (int p = 0; p < KNN_K; ++p) {
                bool sm = dc < bd[p];
                float td = bd[p]; int ti = bi[p];
                bd[p] = sm ? dc : td;  bi[p] = sm ? ic : ti;
                dc    = sm ? td : dc;  ic    = sm ? ti : ic;
            }
        }
    }
#pragma unroll
    for (int p = 0; p < KNN_K; ++p) nbr[(size_t)i * KNN_K + p] = bi[p];
}

// ---------------------------------------------------------------------------
// Kernel 4: per-node gather attention. One wave32 per dst node.
// ---------------------------------------------------------------------------
__device__ __forceinline__ float dot4(float4 a, float4 b) {
    return a.x * b.x + a.y * b.y + a.z * b.z + a.w * b.w;
}
__device__ __forceinline__ float clip5(float x) {
    return fminf(fmaxf(x, -5.f), 5.f);
}

__global__ __launch_bounds__(256)
void attn_gather_kernel(const float* __restrict__ Q,
                        const float* __restrict__ Km,
                        const float* __restrict__ Vm,
                        const float* __restrict__ G,
                        const int* __restrict__ nbr,
                        float* __restrict__ out, int n)
{
    const int wid  = blockIdx.x * (blockDim.x >> 5) + (threadIdx.x >> 5);
    const int lane = threadIdx.x & 31;
    if (wid >= n) return;
    const int i = wid;
    const float inv_scale = 1.0f / 22.62741699796952f;   // 1/sqrt(512)

    const float4* qp = (const float4*)(Q + (size_t)i * NODES_DIM + lane * 16);
    float4 q0 = qp[0], q1 = qp[1], q2 = qp[2], q3 = qp[3];

    float gx = G[3 * i], gy = G[3 * i + 1], gz = G[3 * i + 2];

    float4 a0 = {0,0,0,0}, a1 = {0,0,0,0}, a2 = {0,0,0,0}, a3 = {0,0,0,0};
    float zsum = 0.f;

#pragma unroll
    for (int s = 0; s < KNN_K; ++s) {
        int src = nbr[(size_t)i * KNN_K + s];

        const float4* kp = (const float4*)(Km + (size_t)src * NODES_DIM + lane * 16);
        float dot = dot4(kp[0], q0) + dot4(kp[1], q1) + dot4(kp[2], q2) + dot4(kp[3], q3);
#pragma unroll
        for (int off = 16; off > 0; off >>= 1)
            dot += __shfl_xor(dot, off, 32);

        float score = expf(clip5(dot * inv_scale));

        float dx = G[3 * src]     - gx;
        float dy = G[3 * src + 1] - gy;
        float dz = G[3 * src + 2] - gz;
        float dist = -sqrtf(dx * dx + dy * dy + dz * dz + 1e-6f);
        float w = score * expf(clip5(dist * inv_scale));

        const float4* vp = (const float4*)(Vm + (size_t)src * NODES_DIM + lane * 16);
        float4 v0 = vp[0], v1 = vp[1], v2 = vp[2], v3 = vp[3];
        a0.x += w * v0.x; a0.y += w * v0.y; a0.z += w * v0.z; a0.w += w * v0.w;
        a1.x += w * v1.x; a1.y += w * v1.y; a1.z += w * v1.z; a1.w += w * v1.w;
        a2.x += w * v2.x; a2.y += w * v2.y; a2.z += w * v2.z; a2.w += w * v2.w;
        a3.x += w * v3.x; a3.y += w * v3.y; a3.z += w * v3.z; a3.w += w * v3.w;

        zsum += score;
    }

    float inv = (zsum > 0.f) ? (1.f / zsum) : 1.f;
    float4* op = (float4*)(out + (size_t)i * NODES_DIM + lane * 16);
    op[0] = make_float4(a0.x * inv, a0.y * inv, a0.z * inv, a0.w * inv);
    op[1] = make_float4(a1.x * inv, a1.y * inv, a1.z * inv, a1.w * inv);
    op[2] = make_float4(a2.x * inv, a2.y * inv, a2.z * inv, a2.w * inv);
    op[3] = make_float4(a3.x * inv, a3.y * inv, a3.z * inv, a3.w * inv);
}

// ---------------------------------------------------------------------------
extern "C" void kernel_launch(void* const* d_in, const int* in_sizes, int n_in,
                              void* d_out, int out_size, void* d_ws, size_t ws_size,
                              hipStream_t stream) {
    const float* h  = (const float*)d_in[0];
    const float* Wq = (const float*)d_in[1];
    const float* Wk = (const float*)d_in[2];
    const float* Wv = (const float*)d_in[3];
    const float* Wg = (const float*)d_in[4];
    float* out = (float*)d_out;

    const int n = in_sizes[0] / NODES_DIM;   // 8192
    const size_t nd = (size_t)n * NODES_DIM;
    const size_t wd = (size_t)NODES_DIM * NODES_DIM;

    // workspace layout
    char* ws = (char*)d_ws;
    float* Q  = (float*)ws;  ws += nd * sizeof(float);
    float* Kt = (float*)ws;  ws += nd * sizeof(float);
    float* Vt = (float*)ws;  ws += nd * sizeof(float);
    float* G  = (float*)ws;  ws += (size_t)n * 3 * sizeof(float);
    int*   nb = (int*)ws;    ws += (size_t)n * KNN_K * sizeof(int);
    __bf16* h_hi = (__bf16*)ws; ws += nd * sizeof(__bf16);
    __bf16* h_lo = (__bf16*)ws; ws += nd * sizeof(__bf16);
    __bf16* Wqh  = (__bf16*)ws; ws += wd * sizeof(__bf16);
    __bf16* Wql  = (__bf16*)ws; ws += wd * sizeof(__bf16);
    __bf16* Wkh  = (__bf16*)ws; ws += wd * sizeof(__bf16);
    __bf16* Wkl  = (__bf16*)ws; ws += wd * sizeof(__bf16);
    __bf16* Wvh  = (__bf16*)ws; ws += wd * sizeof(__bf16);
    __bf16* Wvl  = (__bf16*)ws; ws += wd * sizeof(__bf16);

    // 0) bf16 hi/lo splits
    split_bf16_kernel<<<(int)((nd + 255) / 256), 256, 0, stream>>>(h, h_hi, h_lo, (int)nd);
    split_bf16_kernel<<<(int)((wd + 255) / 256), 256, 0, stream>>>(Wq, Wqh, Wql, (int)wd);
    split_bf16_kernel<<<(int)((wd + 255) / 256), 256, 0, stream>>>(Wk, Wkh, Wkl, (int)wd);
    split_bf16_kernel<<<(int)((wd + 255) / 256), 256, 0, stream>>>(Wv, Wvh, Wvl, (int)wd);

    // 1) Q/K/V projections: bf16x3 WMMA, TDM-staged weight tiles in LDS
    dim3 gg(n / BLOCK_M, NODES_DIM / TILE_N, 3);
    size_t gemmLds = (size_t)2 * 64 * LDS_ROW * sizeof(__bf16);   // 133120 B
    qkv_gemm_bf16x3<<<gg, 256, gemmLds, stream>>>(h_hi, h_lo, Wqh, Wql, Wkh, Wkl,
                                                  Wvh, Wvl, Q, Kt, Vt);

    // 2) 3-d coordinate projection
    g_proj_kernel<<<(n + 255) / 256, 256, 0, stream>>>(h, Wg, G, n);

    // 3) KNN (coords LDS-resident)
    size_t ldsBytes = (size_t)3 * n * sizeof(float);
    knn_kernel<<<(n + 255) / 256, 256, ldsBytes, stream>>>(G, nb, n);

    // 4) gather attention + normalize
    attn_gather_kernel<<<(n + 7) / 8, 256, 0, stream>>>(Q, Kt, Vt, G, nb, out, n);
}
